// FeedForwardQuantum_65481071409111
// MI455X (gfx1250) — matrix operand order repaired and checked by reading, not compiled, not run on hardware
//
#include <hip/hip_runtime.h>

// ---------------- problem constants (from reference) ----------------
constexpr int EMBED   = 768;
constexpr int NQ      = 8;
constexpr int TOKENS  = 8 * 2048;          // B*S
constexpr int TOK_WG  = 128;               // tokens per workgroup
constexpr int THREADS = 256;               // 8 wave32
constexpr int KTILE   = 32;                // WMMA K per step (bf16)
constexpr int KCH     = EMBED / KTILE;     // 24
constexpr int XPITCH  = 36;                // f32 tile pitch: 16B-aligned rows, bank-skewed

typedef __attribute__((ext_vector_type(16))) __bf16 v16bf;
typedef __attribute__((ext_vector_type(8)))  float  v8f;

union FragBF { v16bf v; __bf16 h[16]; unsigned short u[16]; };

// ---------------- CDNA5 async global->LDS copy ----------------------
typedef int v4i_ __attribute__((vector_size(16)));
typedef __attribute__((address_space(1))) v4i_ gv4i;   // global (AS1)
typedef __attribute__((address_space(3))) v4i_ lv4i;   // LDS (AS3)

#if defined(__has_builtin) && __has_builtin(__builtin_amdgcn_global_load_async_to_lds_b128)
__device__ __forceinline__ void async_ld_b128(void* lp, const float* gp) {
  __builtin_amdgcn_global_load_async_to_lds_b128((gv4i*)gp, (lv4i*)lp, 0, 0);
}
#else
__device__ __forceinline__ void async_ld_b128(void* lp, const float* gp) {
  // GV mode: vdst = LDS byte offset (low 32 bits of generic LDS addr), vaddr = 64-bit
  unsigned loff = (unsigned)(uintptr_t)lp;
  asm volatile("global_load_async_to_lds_b128 %0, %1, off"
               :: "v"(loff), "v"(gp) : "memory");
}
#endif

__device__ __forceinline__ void wait_async0() {
#if defined(__has_builtin) && __has_builtin(__builtin_amdgcn_s_wait_asynccnt)
  __builtin_amdgcn_s_wait_asynccnt(0);
#else
  asm volatile("s_wait_asynccnt 0x0" ::: "memory");
#endif
}

// pack two float4 into 8 bf16 of a fragment at offset o
__device__ __forceinline__ void pack8(FragBF& f, int o, float4 a, float4 b) {
  f.h[o + 0] = (__bf16)a.x; f.h[o + 1] = (__bf16)a.y;
  f.h[o + 2] = (__bf16)a.z; f.h[o + 3] = (__bf16)a.w;
  f.h[o + 4] = (__bf16)b.x; f.h[o + 5] = (__bf16)b.y;
  f.h[o + 6] = (__bf16)b.z; f.h[o + 7] = (__bf16)b.w;
}

// ---------------- statevector gate helpers --------------------------
// amplitude index = lane*8 + j : bits 0..2 = slot j (in-lane), bits 3..7 = lane.
__device__ inline void rx_gate(float re[8], float im[8], int bit, float c, float s) {
  if (bit < 3) {
    const int m = 1 << bit;
#pragma unroll
    for (int j = 0; j < 8; ++j) {
      if (j & m) continue;
      const int j1 = j | m;
      float r0 = re[j], i0 = im[j], r1 = re[j1], i1 = im[j1];
      re[j]  = c * r0 + s * i1;  im[j]  = c * i0 - s * r1;   // (-i s) off-diag
      re[j1] = c * r1 + s * i0;  im[j1] = c * i1 - s * r0;
    }
  } else {
    const int lm = 1 << (bit - 3);
#pragma unroll
    for (int j = 0; j < 8; ++j) {
      float pr = __shfl_xor(re[j], lm, 32);
      float pi = __shfl_xor(im[j], lm, 32);
      re[j] = c * re[j] + s * pi;
      im[j] = c * im[j] - s * pr;
    }
  }
}

__device__ inline void ry_gate(float re[8], float im[8], int bit, float c, float s, int lane) {
  if (bit < 3) {
    const int m = 1 << bit;
#pragma unroll
    for (int j = 0; j < 8; ++j) {
      if (j & m) continue;
      const int j1 = j | m;
      float r0 = re[j], i0 = im[j], r1 = re[j1], i1 = im[j1];
      re[j]  = c * r0 - s * r1;  im[j]  = c * i0 - s * i1;
      re[j1] = c * r1 + s * r0;  im[j1] = c * i1 + s * i0;
    }
  } else {
    const int lm = 1 << (bit - 3);
    const float sg = (lane & lm) ? s : -s;
#pragma unroll
    for (int j = 0; j < 8; ++j) {
      float pr = __shfl_xor(re[j], lm, 32);
      float pi = __shfl_xor(im[j], lm, 32);
      re[j] = c * re[j] + sg * pr;
      im[j] = c * im[j] + sg * pi;
    }
  }
}

__device__ inline void cnot_gate(float re[8], float im[8], int cbit, int tbit, int lane) {
  if (cbit < 3 && tbit < 3) {                      // both in-lane
    const int cm = 1 << cbit, tm = 1 << tbit;
#pragma unroll
    for (int j = 0; j < 8; ++j) {
      if ((j & cm) && !(j & tm)) {
        const int j1 = j | tm;
        float t;
        t = re[j]; re[j] = re[j1]; re[j1] = t;
        t = im[j]; im[j] = im[j1]; im[j1] = t;
      }
    }
  } else if (cbit >= 3 && tbit < 3) {              // ctrl lane bit, tgt in-lane
    const int clm = 1 << (cbit - 3), tm = 1 << tbit;
    if (lane & clm) {
#pragma unroll
      for (int j = 0; j < 8; ++j) {
        if (!(j & tm)) {
          const int j1 = j | tm;
          float t;
          t = re[j]; re[j] = re[j1]; re[j1] = t;
          t = im[j]; im[j] = im[j1]; im[j1] = t;
        }
      }
    }
  } else if (cbit < 3 && tbit >= 3) {              // ctrl in-lane, tgt lane bit
    const int cm = 1 << cbit, tlm = 1 << (tbit - 3);
#pragma unroll
    for (int j = 0; j < 8; ++j) {
      float pr = __shfl_xor(re[j], tlm, 32);       // shuffles convergent
      float pi = __shfl_xor(im[j], tlm, 32);
      if (j & cm) { re[j] = pr; im[j] = pi; }
    }
  } else {                                         // both lane bits
    const int clm = 1 << (cbit - 3), tlm = 1 << (tbit - 3);
#pragma unroll
    for (int j = 0; j < 8; ++j) {
      float pr = __shfl_xor(re[j], tlm, 32);
      float pi = __shfl_xor(im[j], tlm, 32);
      if (lane & clm) { re[j] = pr; im[j] = pi; }
    }
  }
}

// ---------------- fused kernel --------------------------------------
__global__ __launch_bounds__(THREADS)
void ffq_fused(const float* __restrict__ x,  const float* __restrict__ W1,
               const float* __restrict__ b1, const float* __restrict__ theta,
               const float* __restrict__ W2, const float* __restrict__ b2,
               float* __restrict__ out)
{
  __shared__ __align__(16) float xt[2][TOK_WG * XPITCH];  // double-buffered f32 x tile
  __shared__ __align__(16) float qin[TOK_WG * NQ];        // RX encode angles
  __shared__ __align__(16) float zt [TOK_WG * NQ];        // PauliZ expectations

  const int tid  = threadIdx.x;
  const int lane = tid & 31;
  const int wave = tid >> 5;
  const int hi   = lane >> 4;                      // half-wave 0/1
  const int l16  = lane & 15;
  const int gt0  = blockIdx.x * TOK_WG;            // first token of this WG
  const int wt0  = wave * 16;                      // this wave's 16-token slice

  // tile-copy geometry: 128 rows x 32 f32 = 1024 float4; 4 per thread
  const float* xbase = x + (size_t)gt0 * EMBED;

  // ============ Phase 1: q_in = x @ W1^T + b1 via bf16 WMMA ============
  // prologue: async-copy tile 0 into buffer 0
#pragma unroll
  for (int i = 0; i < 4; ++i) {
    const int v = tid + i * THREADS;
    const int row = v >> 3, c4 = (v & 7) * 4;
    async_ld_b128(&xt[0][row * XPITCH + c4],
                  xbase + (size_t)row * EMBED + c4);
  }
  wait_async0();
  __syncthreads();

  v8f acc = {};
  for (int kc = 0; kc < KCH; ++kc) {
    const int cur = kc & 1;
    // overlap: issue async copy of tile kc+1 into the other buffer
    if (kc + 1 < KCH) {
      const int nxt = cur ^ 1;
#pragma unroll
      for (int i = 0; i < 4; ++i) {
        const int v = tid + i * THREADS;
        const int row = v >> 3, c4 = (v & 7) * 4;
        async_ld_b128(&xt[nxt][row * XPITCH + c4],
                      xbase + (size_t)row * EMBED + (kc + 1) * KTILE + c4);
      }
    }

    // A fragment (16x32 bf16): row M = l16, K = hi*16..+15
    FragBF a;
    {
      const float* rp = &xt[cur][(wt0 + l16) * XPITCH + hi * 16];
      pack8(a, 0, *(const float4*)(rp),      *(const float4*)(rp + 4));
      pack8(a, 8, *(const float4*)(rp + 8),  *(const float4*)(rp + 12));
    }
    // B fragment (32x16 bf16): col N = l16 (zero-padded q>=8), K = hi*16..+15
    FragBF b;
#pragma unroll
    for (int k = 0; k < 16; ++k) b.u[k] = 0;
    if (l16 < NQ) {
      const float4* w1r = (const float4*)(W1 + (size_t)l16 * EMBED + kc * KTILE + hi * 16);
      pack8(b, 0, w1r[0], w1r[1]);
      pack8(b, 8, w1r[2], w1r[3]);
    }
    acc = __builtin_amdgcn_wmma_f32_16x16x32_bf16(false, a.v, false, b.v,
                                                  (short)0, acc, false, false);
    wait_async0();      // my async copies for tile kc+1 done
    __syncthreads();    // everyone done reading buf[cur] & writing buf[nxt]
  }
  // scatter q_in (+b1): C element (M = r + 8*hi, N = l16)
  if (l16 < NQ) {
    const float bb = b1[l16];
#pragma unroll
    for (int r = 0; r < 8; ++r)
      qin[(wt0 + r + 8 * hi) * NQ + l16] = acc[r] + bb;
  }
  __syncthreads();

  // ============ Phase 2: 8-qubit statevector sim, one token per wave ===
  float cy[NQ], sy[NQ];
#pragma unroll
  for (int w = 0; w < NQ; ++w)
    __sincosf(theta[w] * 0.5f, &sy[w], &cy[w]);

  for (int t = 0; t < 16; ++t) {
    const int tok = wt0 + t;
    float4 a0 = *(const float4*)&qin[tok * NQ];
    float4 a1 = *(const float4*)&qin[tok * NQ + 4];
    const float ang[NQ] = {a0.x, a0.y, a0.z, a0.w, a1.x, a1.y, a1.z, a1.w};
    float cx[NQ], sx[NQ];
#pragma unroll
    for (int w = 0; w < NQ; ++w)
      __sincosf(ang[w] * 0.5f, &sx[w], &cx[w]);

    float re[8], im[8];
#pragma unroll
    for (int j = 0; j < 8; ++j) { re[j] = 0.f; im[j] = 0.f; }
    if (lane == 0) re[0] = 1.f;

    // wire w acts on index bit (7-w)
#pragma unroll
    for (int w = 0; w < NQ; ++w) rx_gate(re, im, 7 - w, cx[w], sx[w]);
#pragma unroll
    for (int w = 0; w < NQ; ++w) ry_gate(re, im, 7 - w, cy[w], sy[w], lane);
    // CNOT ring (N_QLAYERS-1 == 1 pass)
#pragma unroll
    for (int w = 0; w < NQ - 1; ++w) cnot_gate(re, im, 7 - w, 6 - w, lane);
    cnot_gate(re, im, 0, 7, lane);

    // PauliZ expectations per wire
    float p[8];
#pragma unroll
    for (int j = 0; j < 8; ++j) p[j] = re[j] * re[j] + im[j] * im[j];
#pragma unroll
    for (int w = 0; w < NQ; ++w) {
      const int bit = 7 - w;
      float a2;
      if (bit < 3) {
        a2 = 0.f;
#pragma unroll
        for (int j = 0; j < 8; ++j) a2 += ((j >> bit) & 1) ? -p[j] : p[j];
      } else {
        float s = 0.f;
#pragma unroll
        for (int j = 0; j < 8; ++j) s += p[j];
        a2 = ((lane >> (bit - 3)) & 1) ? -s : s;
      }
#pragma unroll
      for (int off = 16; off > 0; off >>= 1)
        a2 += __shfl_xor(a2, off, 32);
      if (lane == 0) zt[tok * NQ + w] = a2;
    }
  }
  __syncthreads();

  // ============ Phase 3: out = relu(z @ W2^T + b2) via bf16 WMMA =======
  // A fragment: z tile 16x32 (K 8..31 zero => half-wave 1 entirely zero)
  FragBF az;
#pragma unroll
  for (int k = 0; k < 16; ++k) az.u[k] = 0;
  if (hi == 0) {
    const float4* zp = (const float4*)&zt[(wt0 + l16) * NQ];
    pack8(az, 0, zp[0], zp[1]);
  }

  for (int e0 = 0; e0 < EMBED; e0 += 16) {
    FragBF bw;                                     // B: col N = e0+l16, K<8 = W2 row
#pragma unroll
    for (int k = 0; k < 16; ++k) bw.u[k] = 0;
    if (hi == 0) {
      const float4* w2r = (const float4*)(W2 + (size_t)(e0 + l16) * NQ);
      pack8(bw, 0, w2r[0], w2r[1]);
    }
    v8f co = {};
    co = __builtin_amdgcn_wmma_f32_16x16x32_bf16(false, az.v, false, bw.v,
                                                 (short)0, co, false, false);
    const float bb = b2[e0 + l16];
#pragma unroll
    for (int r = 0; r < 8; ++r) {
      const int m = r + 8 * hi;
      float v = co[r] + bb;
      out[(size_t)(gt0 + wt0 + m) * EMBED + e0 + l16] = v > 0.f ? v : 0.f;
    }
  }
}

// ---------------- harness entry -------------------------------------
extern "C" void kernel_launch(void* const* d_in, const int* in_sizes, int n_in,
                              void* d_out, int out_size, void* d_ws, size_t ws_size,
                              hipStream_t stream) {
  const float* x     = (const float*)d_in[0];
  const float* W1    = (const float*)d_in[1];
  const float* b1    = (const float*)d_in[2];
  const float* theta = (const float*)d_in[3];
  const float* W2    = (const float*)d_in[4];
  const float* b2    = (const float*)d_in[5];
  float* out         = (float*)d_out;

  dim3 grid(TOKENS / TOK_WG);   // 128 workgroups
  dim3 block(THREADS);          // 8 wave32 each
  hipLaunchKernelGGL(ffq_fused, grid, block, 0, stream,
                     x, W1, b1, theta, W2, b2, out);
}
